// DGCNN_semseg_55276229099976
// MI455X (gfx1250) — matrix operand order repaired and checked by reading, not compiled, not run on hardware
//
#include <hip/hip_runtime.h>
#include <hip/hip_bf16.h>

typedef __attribute__((ext_vector_type(16))) _Float16 v16h;
typedef __attribute__((ext_vector_type(8)))  float    v8f;
typedef __attribute__((ext_vector_type(4)))  unsigned int u32x4;
typedef __attribute__((ext_vector_type(8)))  int      i32x8;
typedef __attribute__((ext_vector_type(4)))  int      i32x4;

#define NPTS  4096
#define BATCH 4
#define KNN   40
#define FNEG  (-3.4e38f)

// ---------------- WMMA helpers (CDNA5 16x16x32 f16 layouts) ----------------

__device__ __forceinline__ v8f wmma32(v16h a, v16h b, v8f c) {
  // 8 args: (neg_a, A, neg_b, B, c_mod, C, reuse_a, reuse_b)
  return __builtin_amdgcn_wmma_f32_16x16x32_f16(false, a, false, b, (short)0, c,
                                                false, false);
}

// A matrix 16x32 (MxK), row-major weights. Layout: lane<16 holds row
// M=lane, K=0..15; lane>=16 holds row M=lane-16, K=16..31.
__device__ __forceinline__ v16h load_a16(const _Float16* w, int ld, int m0,
                                         int k0, int lane) {
  const int h = (lane >> 4) & 1;
  const int r = lane & 15;
  const _Float16* p = w + (size_t)(m0 + r) * ld + k0;
  v16h a;
#pragma unroll
  for (int i = 0; i < 16; ++i) a[i] = p[16 * h + i];
  return a;
}

// B matrix 32x16 (KxN), column-major storage. Lane<16: N=lane, K=k0..k0+15;
// lane>=16: N=lane-16, K=k0+16..k0+31 (16 contiguous halves each).
__device__ __forceinline__ v16h load_b16(const _Float16* colbase, int k0, int h) {
  const _Float16* p = colbase + k0 + 16 * h;
  v16h b;
#pragma unroll
  for (int i = 0; i < 16; ++i) b[i] = p[i];
  return b;
}

__device__ __forceinline__ float lrelu(float v) { return v > 0.f ? v : 0.2f * v; }

__device__ __forceinline__ void atomicMaxF(float* addr, float val) {
  unsigned int* ua = (unsigned int*)addr;
  unsigned int old = *ua;
  while (__uint_as_float(old) < val) {
    unsigned int assumed = old;
    old = atomicCAS(ua, assumed, __float_as_uint(val));
    if (old == assumed) break;
  }
}

// ---------------- TDM gather: 16 rows-by-index -> LDS (ISA ch.8 D#) --------
// Gather-mode descriptor: 2D, data_size=2B, row width 64 halves, row stride
// `sld` halves, 16x 16-bit row indices in groups 2/3. Issued by one wave,
// tracked with TENSORcnt. This toolchain's builtin takes 6 args (extra
// int32x8 SGPR group; unused here -> zeros).
__device__ __forceinline__ void tdm_gather16(unsigned lds_addr,
                                             const _Float16* gbase, int sld,
                                             int jv /* per-lane row index */) {
  unsigned long long ga = (unsigned long long)(size_t)gbase;
  u32x4 g0;
  g0[0] = 0x80000001u;                       // count=1 | gather_mode | idx16
  g0[1] = lds_addr;                          // lds_addr [63:32]
  g0[2] = (unsigned)(ga & 0xffffffffu);      // global_addr lo
  g0[3] = (unsigned)((ga >> 32) & 0x01ffffffu) | 0x80000000u;  // hi | type=2
  i32x8 g1;
  g1[0] = 0x00010000;                        // data_size = 2 bytes
  g1[1] = (64 << 16);                        // tensor_dim0 = 64 (bit48)
  g1[2] = (int)(4096u << 16);                // tensor_dim1 = NPTS (bit80)
  g1[3] = (64 << 16);                        // tile_dim0 = 64 (bit112)
  g1[4] = 16;                                // tile_dim1 = #valid indices
  g1[5] = sld;                               // tensor_dim0_stride
  g1[6] = 0;
  g1[7] = 0;
  unsigned wl[8];
#pragma unroll
  for (int i = 0; i < 8; ++i) {
    unsigned lo = (unsigned)__builtin_amdgcn_readlane(jv, 2 * i) & 0xffffu;
    unsigned hi = (unsigned)__builtin_amdgcn_readlane(jv, 2 * i + 1) & 0xffffu;
    wl[i] = lo | (hi << 16);
  }
  i32x4 g2; g2[0] = (int)wl[0]; g2[1] = (int)wl[1]; g2[2] = (int)wl[2]; g2[3] = (int)wl[3];
  i32x4 g3; g3[0] = (int)wl[4]; g3[1] = (int)wl[5]; g3[2] = (int)wl[6]; g3[3] = (int)wl[7];
  i32x8 gz = {};                             // unused 5th SGPR group
  __builtin_amdgcn_tensor_load_to_lds(g0, g1, g2, g3, gz, 0);
}

// ---------------- prep kernels ----------------

__global__ void prep_layer(const float* w, const float* g, const float* bb,
                           const float* m, const float* v, int O, int I,
                           int Ipad, _Float16* wh, float* scale, float* shift) {
  int tid = blockIdx.x * blockDim.x + threadIdx.x;
  if (tid < O * Ipad) {
    int o = tid / Ipad, i = tid % Ipad;
    wh[tid] = (i < I) ? (_Float16)w[o * I + i] : (_Float16)0.0f;
  }
  if (tid < O) {
    float s = g[tid] * rsqrtf(v[tid] + 1e-5f);
    scale[tid] = s;
    shift[tid] = bb[tid] - m[tid] * s;
  }
}

__global__ void cvt_f16(const float* src, _Float16* dst, int n) {
  int i = blockIdx.x * blockDim.x + threadIdx.x;
  if (i < n) dst[i] = (_Float16)src[i];
}

__global__ void init_neg(float* g, int n) {
  int i = blockIdx.x * blockDim.x + threadIdx.x;
  if (i < n) g[i] = FNEG;
}

// ---------------- knn ----------------

__global__ void knn_l1_kernel(const float* x, int* idx) {
  int t = blockIdx.x * blockDim.x + threadIdx.x;
  if (t >= BATCH * NPTS) return;
  int b = t / NPTS, n = t % NPTS;
  const float* xb = x + (size_t)b * NPTS * 14;
  float a0 = xb[n * 14 + 0], c0 = xb[n * 14 + 2];
  float bd[KNN]; int bi[KNN];
#pragma unroll
  for (int i = 0; i < KNN; ++i) { bd[i] = 3.4e38f; bi[i] = 0; }
  for (int m = 0; m < NPTS; ++m) {
    float d = fabsf(a0 - xb[m * 14]) + fabsf(c0 - xb[m * 14 + 2]);
    if (d < bd[KNN - 1]) {
      int j = KNN - 1;
      while (j > 0 && bd[j - 1] > d) { bd[j] = bd[j - 1]; bi[j] = bi[j - 1]; --j; }
      bd[j] = d; bi[j] = m;
    }
  }
  int* op = idx + (size_t)t * KNN;
  for (int i = 0; i < KNN; ++i) op[i] = bi[i];
}

__global__ void knn_l2_kernel(const _Float16* cath, int coff, int* idx) {
  int t = blockIdx.x * blockDim.x + threadIdx.x;
  if (t >= BATCH * NPTS) return;
  int b = t / NPTS, n = t % NPTS;
  const _Float16* me = cath + ((size_t)(b * NPTS) + n) * 192 + coff;
  float mine[64];
#pragma unroll
  for (int c = 0; c < 64; ++c) mine[c] = (float)me[c];
  float bd[KNN]; int bi[KNN];
#pragma unroll
  for (int i = 0; i < KNN; ++i) { bd[i] = 3.4e38f; bi[i] = 0; }
  for (int m = 0; m < NPTS; ++m) {
    const _Float16* q = cath + ((size_t)(b * NPTS) + m) * 192 + coff;
    float d = 0.f;
#pragma unroll
    for (int c = 0; c < 64; ++c) { float df = mine[c] - (float)q[c]; d += df * df; }
    if (d < bd[KNN - 1]) {
      int j = KNN - 1;
      while (j > 0 && bd[j - 1] > d) { bd[j] = bd[j - 1]; bi[j] = bi[j - 1]; --j; }
      bd[j] = d; bi[j] = m;
    }
  }
  int* op = idx + (size_t)t * KNN;
  for (int i = 0; i < KNN; ++i) op[i] = bi[i];
}

// ---------------- edge-conv, first block (CIN=14, VALU gather) -------------

template <int CIN, int CPAD>
__global__ __launch_bounds__(128) void edgeconv2_kernel(
    const _Float16* src, int sld, int soff, const int* idx,
    const _Float16* w1, const float* s1, const float* t1,
    const _Float16* w2, const float* s2, const float* t2,
    _Float16* dst, int dld, int doff) {
  __shared__ _Float16 feat[16][CPAD];
  __shared__ _Float16 hmid[16][64];
  const int b = blockIdx.y;
  const int p0 = blockIdx.x * 16;
  const int tid = threadIdx.x;
  const int lane = tid & 31, wave = tid >> 5;
  const int h = lane >> 4, col = lane & 15;
  const int m0 = wave * 16;

  float runmax[8];
#pragma unroll
  for (int r = 0; r < 8; ++r) runmax[r] = FNEG;

  for (int t = 0; t < KNN; ++t) {
    for (int e = tid; e < 16 * CPAD; e += 128) {
      int cc = e % CPAD, cl = e / CPAD;
      int np = p0 + cl;
      int j = idx[((size_t)(b * NPTS) + np) * KNN + t];
      const _Float16* ctr = src + ((size_t)(b * NPTS) + np) * sld + soff;
      const _Float16* nbr = src + ((size_t)(b * NPTS) + j) * sld + soff;
      float val;
      if (cc < CIN)            val = (float)nbr[cc] - (float)ctr[cc];
      else if (cc < 2 * CIN)   val = (float)ctr[cc - CIN];
      else                     val = 0.0f;
      feat[cl][cc] = (_Float16)val;
    }
    __syncthreads();

    v8f acc = {};
#pragma unroll
    for (int ks = 0; ks < CPAD / 32; ++ks) {
      v16h a = load_a16(w1, CPAD, m0, ks * 32, lane);
      v16h bm = load_b16(&feat[col][0], ks * 32, h);
      acc = wmma32(a, bm, acc);
    }
#pragma unroll
    for (int r = 0; r < 8; ++r) {
      int m = m0 + 8 * h + r;
      hmid[col][m] = (_Float16)lrelu(acc[r] * s1[m] + t1[m]);
    }
    __syncthreads();

    v8f acc2 = {};
#pragma unroll
    for (int ks = 0; ks < 2; ++ks) {
      v16h a = load_a16(w2, 64, m0, ks * 32, lane);
      v16h bm = load_b16(&hmid[col][0], ks * 32, h);
      acc2 = wmma32(a, bm, acc2);
    }
#pragma unroll
    for (int r = 0; r < 8; ++r) {
      int m = m0 + 8 * h + r;
      runmax[r] = fmaxf(runmax[r], lrelu(acc2[r] * s2[m] + t2[m]));
    }
    __syncthreads();
  }
#pragma unroll
  for (int r = 0; r < 8; ++r) {
    int m = m0 + 8 * h + r;
    dst[((size_t)(b * NPTS) + p0 + col) * dld + doff + m] = (_Float16)runmax[r];
  }
}

// ---------------- edge-conv, 64-ch blocks with TDM-gathered neighbors ------
// Double-buffered: wave 0 issues the TDM index-gather for neighbor slot t+1
// while all waves run WMMA on slot t; sync via s_wait_tensorcnt + barrier.

__device__ __forceinline__ v16h edge_b(const _Float16 (*nb)[64],
                                       const _Float16 (*ct)[64], int col,
                                       int k0, int h) {
  int base = k0 + 16 * h;
  v16h bm;
  if (base < 64) {
#pragma unroll
    for (int i = 0; i < 16; ++i)
      bm[i] = (_Float16)(nb[col][base + i] - ct[col][base + i]);
  } else {
#pragma unroll
    for (int i = 0; i < 16; ++i) bm[i] = ct[col][base - 64 + i];
  }
  return bm;
}

__global__ __launch_bounds__(128) void edgeconv2_tdm(
    const _Float16* src, int sld, int soff, const int* idx,
    const _Float16* w1, const float* s1, const float* t1,
    const _Float16* w2, const float* s2, const float* t2,
    _Float16* dst, int dld, int doff) {
  __shared__ _Float16 nbr[2][16][64];
  __shared__ _Float16 ctr[16][64];
  __shared__ _Float16 hmid[16][64];
  const int b = blockIdx.y, p0 = blockIdx.x * 16;
  const int tid = threadIdx.x, lane = tid & 31, wave = tid >> 5;
  const int h = lane >> 4, col = lane & 15;
  const int m0 = wave * 16;

  for (int e = tid; e < 16 * 64; e += 128) {
    int cl = e >> 6, c = e & 63;
    ctr[cl][c] = src[((size_t)(b * NPTS) + p0 + cl) * sld + soff + c];
  }
  const _Float16* gbase = src + (size_t)(b * NPTS) * sld + soff;
  if (wave == 0) {
    int j0 = idx[((size_t)(b * NPTS) + p0 + col) * KNN + 0];
    tdm_gather16((unsigned)(size_t)&nbr[0][0][0], gbase, sld, j0);
  }
  __syncthreads();

  float runmax[8];
#pragma unroll
  for (int r = 0; r < 8; ++r) runmax[r] = FNEG;

  for (int t = 0; t < KNN; ++t) {
    if (wave == 0) __builtin_amdgcn_s_wait_tensorcnt(0);
    __syncthreads();  // gather for slot t landed in LDS
    if (t + 1 < KNN && wave == 0) {
      int jn = idx[((size_t)(b * NPTS) + p0 + col) * KNN + (t + 1)];
      tdm_gather16((unsigned)(size_t)&nbr[(t + 1) & 1][0][0], gbase, sld, jn);
    }
    const _Float16(*nb)[64] = nbr[t & 1];

    v8f acc = {};
#pragma unroll
    for (int ks = 0; ks < 4; ++ks) {
      v16h a = load_a16(w1, 128, m0, ks * 32, lane);
      v16h bm = edge_b(nb, ctr, col, ks * 32, h);
      acc = wmma32(a, bm, acc);
    }
#pragma unroll
    for (int r = 0; r < 8; ++r) {
      int m = m0 + 8 * h + r;
      hmid[col][m] = (_Float16)lrelu(acc[r] * s1[m] + t1[m]);
    }
    __syncthreads();

    v8f acc2 = {};
#pragma unroll
    for (int ks = 0; ks < 2; ++ks) {
      v16h a = load_a16(w2, 64, m0, ks * 32, lane);
      v16h bm = load_b16(&hmid[col][0], ks * 32, h);
      acc2 = wmma32(a, bm, acc2);
    }
#pragma unroll
    for (int r = 0; r < 8; ++r) {
      int m = m0 + 8 * h + r;
      runmax[r] = fmaxf(runmax[r], lrelu(acc2[r] * s2[m] + t2[m]));
    }
    __syncthreads();
  }
#pragma unroll
  for (int r = 0; r < 8; ++r) {
    int m = m0 + 8 * h + r;
    dst[((size_t)(b * NPTS) + p0 + col) * dld + doff + m] = (_Float16)runmax[r];
  }
}

__global__ __launch_bounds__(128) void edgeconv1_tdm(
    const _Float16* src, int sld, int soff, const int* idx,
    const _Float16* w1, const float* s1, const float* t1,
    _Float16* dst, int dld, int doff) {
  __shared__ _Float16 nbr[2][16][64];
  __shared__ _Float16 ctr[16][64];
  const int b = blockIdx.y, p0 = blockIdx.x * 16;
  const int tid = threadIdx.x, lane = tid & 31, wave = tid >> 5;
  const int h = lane >> 4, col = lane & 15;
  const int m0 = wave * 16;

  for (int e = tid; e < 16 * 64; e += 128) {
    int cl = e >> 6, c = e & 63;
    ctr[cl][c] = src[((size_t)(b * NPTS) + p0 + cl) * sld + soff + c];
  }
  const _Float16* gbase = src + (size_t)(b * NPTS) * sld + soff;
  if (wave == 0) {
    int j0 = idx[((size_t)(b * NPTS) + p0 + col) * KNN + 0];
    tdm_gather16((unsigned)(size_t)&nbr[0][0][0], gbase, sld, j0);
  }
  __syncthreads();

  float runmax[8];
#pragma unroll
  for (int r = 0; r < 8; ++r) runmax[r] = FNEG;

  for (int t = 0; t < KNN; ++t) {
    if (wave == 0) __builtin_amdgcn_s_wait_tensorcnt(0);
    __syncthreads();
    if (t + 1 < KNN && wave == 0) {
      int jn = idx[((size_t)(b * NPTS) + p0 + col) * KNN + (t + 1)];
      tdm_gather16((unsigned)(size_t)&nbr[(t + 1) & 1][0][0], gbase, sld, jn);
    }
    const _Float16(*nb)[64] = nbr[t & 1];

    v8f acc = {};
#pragma unroll
    for (int ks = 0; ks < 4; ++ks) {
      v16h a = load_a16(w1, 128, m0, ks * 32, lane);
      v16h bm = edge_b(nb, ctr, col, ks * 32, h);
      acc = wmma32(a, bm, acc);
    }
#pragma unroll
    for (int r = 0; r < 8; ++r) {
      int m = m0 + 8 * h + r;
      runmax[r] = fmaxf(runmax[r], lrelu(acc[r] * s1[m] + t1[m]));
    }
    __syncthreads();
  }
#pragma unroll
  for (int r = 0; r < 8; ++r) {
    int m = m0 + 8 * h + r;
    dst[((size_t)(b * NPTS) + p0 + col) * dld + doff + m] = (_Float16)runmax[r];
  }
}

// ---------------- pointwise GEMM stages (16x64 per wave, A reuse x4) -------

__global__ __launch_bounds__(128) void conv6_kernel(const _Float16* cath,
                                                    const _Float16* w,
                                                    const float* sc,
                                                    const float* sh, float* g) {
  const int lane = threadIdx.x & 31, wave = threadIdx.x >> 5;
  const int h = lane >> 4, col = lane & 15;
  const int m0 = blockIdx.y * 64 + wave * 16;
  const int colbase = blockIdx.x * 64;
  const int b = colbase >> 12;
  const _Float16* cps[4];
#pragma unroll
  for (int s = 0; s < 4; ++s)
    cps[s] = cath + (size_t)(colbase + s * 16 + col) * 192;
  v8f acc[4] = {};
#pragma unroll
  for (int ks = 0; ks < 6; ++ks) {
    v16h a = load_a16(w, 192, m0, ks * 32, lane);
#pragma unroll
    for (int s = 0; s < 4; ++s) {
      v16h bm = load_b16(cps[s], ks * 32, h);
      acc[s] = wmma32(a, bm, acc[s]);
    }
  }
#pragma unroll
  for (int s = 0; s < 4; ++s) {
#pragma unroll
    for (int r = 0; r < 8; ++r) {
      int m = m0 + 8 * h + r;
      float v = lrelu(acc[s][r] * sc[m] + sh[m]);
#pragma unroll
      for (int mask = 1; mask < 16; mask <<= 1)
        v = fmaxf(v, __shfl_xor(v, mask, 32));
      if (col == 0) atomicMaxF(&g[b * 1024 + m], v);
    }
  }
}

__global__ __launch_bounds__(128) void conv7_kernel(
    const _Float16* cath, const _Float16* gh, const _Float16* w,
    const float* sc, const float* sh, _Float16* h7) {
  const int lane = threadIdx.x & 31, wave = threadIdx.x >> 5;
  const int h = lane >> 4, col = lane & 15;
  const int m0 = blockIdx.y * 64 + wave * 16;
  const int colbase = blockIdx.x * 64;
  const int b = colbase >> 12;
  const _Float16* cps[4];
#pragma unroll
  for (int s = 0; s < 4; ++s)
    cps[s] = cath + (size_t)(colbase + s * 16 + col) * 192;
  const _Float16* gp = gh + b * 1024;
  v8f acc[4] = {};
  for (int ks = 0; ks < 38; ++ks) {
    int k0 = ks * 32;
    __builtin_prefetch(w + (size_t)(m0 + (lane & 15)) * 1216 + k0 + 32, 0, 1);
    v16h a = load_a16(w, 1216, m0, k0, lane);
    if (k0 < 1024) {
      v16h bg = load_b16(gp, k0, h);  // broadcast column: shared by all tiles
#pragma unroll
      for (int s = 0; s < 4; ++s) acc[s] = wmma32(a, bg, acc[s]);
    } else {
#pragma unroll
      for (int s = 0; s < 4; ++s) {
        v16h bm = load_b16(cps[s], k0 - 1024, h);
        acc[s] = wmma32(a, bm, acc[s]);
      }
    }
  }
#pragma unroll
  for (int s = 0; s < 4; ++s) {
#pragma unroll
    for (int r = 0; r < 8; ++r) {
      int m = m0 + 8 * h + r;
      int colg = colbase + s * 16 + col;
      h7[(size_t)colg * 512 + m] = (_Float16)lrelu(acc[s][r] * sc[m] + sh[m]);
    }
  }
}

__global__ __launch_bounds__(128) void conv8_kernel(const _Float16* h7,
                                                    const _Float16* w,
                                                    const float* sc,
                                                    const float* sh,
                                                    float* out) {
  const int lane = threadIdx.x & 31, wave = threadIdx.x >> 5;
  const int h = lane >> 4, col = lane & 15;
  const int m0 = blockIdx.y * 64 + wave * 16;
  const int colbase = blockIdx.x * 64;
  const int b = colbase >> 12;
  const _Float16* cps[4];
#pragma unroll
  for (int s = 0; s < 4; ++s)
    cps[s] = h7 + (size_t)(colbase + s * 16 + col) * 512;
  v8f acc[4] = {};
#pragma unroll
  for (int ks = 0; ks < 16; ++ks) {
    v16h a = load_a16(w, 512, m0, ks * 32, lane);
#pragma unroll
    for (int s = 0; s < 4; ++s) {
      v16h bm = load_b16(cps[s], ks * 32, h);
      acc[s] = wmma32(a, bm, acc[s]);
    }
  }
#pragma unroll
  for (int s = 0; s < 4; ++s) {
#pragma unroll
    for (int r = 0; r < 8; ++r) {
      int m = m0 + 8 * h + r;
      int colg = colbase + s * 16 + col;
      int n = colg & 4095;
      out[((size_t)(b * 256) + m) * NPTS + n] = lrelu(acc[s][r] * sc[m] + sh[m]);
    }
  }
}

// ---------------- launcher ----------------

extern "C" void kernel_launch(void* const* d_in, const int* in_sizes, int n_in,
                              void* d_out, int out_size, void* d_ws,
                              size_t ws_size, hipStream_t stream) {
  (void)in_sizes; (void)n_in; (void)out_size; (void)ws_size;
  const float* x = (const float*)d_in[0];

  static const int O[8]    = {64, 64, 64, 64, 64, 1024, 512, 256};
  static const int I[8]    = {28, 64, 128, 64, 128, 192, 1216, 512};
  static const int Ipad[8] = {32, 64, 128, 64, 128, 192, 1216, 512};

  char* p = (char*)d_ws;
  auto carve = [&](size_t bytes) -> char* {
    char* r = p;
    p += (bytes + 255) & ~(size_t)255;
    return r;
  };
  _Float16* xh   = (_Float16*)carve((size_t)BATCH * NPTS * 14 * 2);
  int*      idx  = (int*)carve((size_t)BATCH * NPTS * KNN * 4);
  _Float16* cath = (_Float16*)carve((size_t)BATCH * NPTS * 192 * 2);
  _Float16* h7   = (_Float16*)carve((size_t)BATCH * NPTS * 512 * 2);
  float*    gbuf = (float*)carve((size_t)BATCH * 1024 * 4);
  _Float16* gh   = (_Float16*)carve((size_t)BATCH * 1024 * 2);
  _Float16* wh[8]; float* sc[8]; float* sh[8];
  for (int l = 0; l < 8; ++l) {
    wh[l] = (_Float16*)carve((size_t)O[l] * Ipad[l] * 2);
    sc[l] = (float*)carve((size_t)O[l] * 4);
    sh[l] = (float*)carve((size_t)O[l] * 4);
  }

  for (int l = 0; l < 8; ++l) {
    const float* w = (const float*)d_in[1 + 5 * l + 0];
    const float* g = (const float*)d_in[1 + 5 * l + 1];
    const float* b = (const float*)d_in[1 + 5 * l + 2];
    const float* m = (const float*)d_in[1 + 5 * l + 3];
    const float* v = (const float*)d_in[1 + 5 * l + 4];
    int tot = O[l] * Ipad[l];
    prep_layer<<<(tot + 255) / 256, 256, 0, stream>>>(w, g, b, m, v, O[l], I[l],
                                                      Ipad[l], wh[l], sc[l], sh[l]);
  }
  cvt_f16<<<(BATCH * NPTS * 14 + 255) / 256, 256, 0, stream>>>(
      x, xh, BATCH * NPTS * 14);

  const int nthr = BATCH * NPTS;
  knn_l1_kernel<<<(nthr + 127) / 128, 128, 0, stream>>>(x, idx);

  dim3 eg(NPTS / 16, BATCH);
  edgeconv2_kernel<14, 32><<<eg, 128, 0, stream>>>(
      xh, 14, 0, idx, wh[0], sc[0], sh[0], wh[1], sc[1], sh[1], cath, 192, 0);
  knn_l2_kernel<<<(nthr + 127) / 128, 128, 0, stream>>>(cath, 0, idx);
  edgeconv2_tdm<<<eg, 128, 0, stream>>>(
      cath, 192, 0, idx, wh[2], sc[2], sh[2], wh[3], sc[3], sh[3], cath, 192, 64);
  knn_l2_kernel<<<(nthr + 127) / 128, 128, 0, stream>>>(cath, 64, idx);
  edgeconv1_tdm<<<eg, 128, 0, stream>>>(
      cath, 192, 64, idx, wh[4], sc[4], sh[4], cath, 192, 128);

  init_neg<<<(BATCH * 1024 + 255) / 256, 256, 0, stream>>>(gbuf, BATCH * 1024);
  conv6_kernel<<<dim3(nthr / 64, 16), 128, 0, stream>>>(cath, wh[5], sc[5],
                                                        sh[5], gbuf);
  cvt_f16<<<(BATCH * 1024 + 255) / 256, 256, 0, stream>>>(gbuf, gh,
                                                          BATCH * 1024);
  conv7_kernel<<<dim3(nthr / 64, 8), 128, 0, stream>>>(cath, gh, wh[6], sc[6],
                                                       sh[6], h7);
  conv8_kernel<<<dim3(nthr / 64, 4), 128, 0, stream>>>(h7, wh[7], sc[7], sh[7],
                                                       (float*)d_out);
}
